// Model_13142599925848
// MI455X (gfx1250) — compile-verified, matrix operand order checked by reading, not run on hardware
//
#include <hip/hip_runtime.h>
#include <hip/hip_bf16.h>

typedef _Float16 v16h __attribute__((ext_vector_type(16)));
typedef float    v8f  __attribute__((ext_vector_type(8)));

#define D64 64

// ---------------------------------------------------------------------------
// Degree counting: cnt[dst[e]] += 1   (cnt buffer holds int bit patterns)
// ---------------------------------------------------------------------------
__global__ void count_kernel(const int* __restrict__ dst, int* __restrict__ cnt, int E) {
    int tid = blockIdx.x * blockDim.x + threadIdx.x;
    if (tid < E) atomicAdd(&cnt[dst[tid]], 1);
}

// In-place int count -> float reciprocal of max(cnt,1)
__global__ void inv_kernel(float* __restrict__ buf, int n) {
    int tid = blockIdx.x * blockDim.x + threadIdx.x;
    if (tid < n) {
        int c = ((const int*)buf)[tid];
        buf[tid] = 1.0f / (float)(c > 1 ? c : 1);
    }
}

// ---------------------------------------------------------------------------
// Weight prep: convert f32 weights to f16 WMMA B-fragments once.
// B[k][n] = W[n][k].  Fragment (nt, kc): lane holds column n = nt*16 + (lane&15),
// halves j -> k = kc*32 + (lane>>4)*8 + j + (j>=8 ? 8 : 0).
// Storage: out[((nt*KC + kc)*32 + lane)*16 + j]  -> each lane's 16 halves are a
// contiguous 32B chunk; a wave's fragment is 1KB contiguous (coalesced b128s).
// ---------------------------------------------------------------------------
__global__ void prep_cat_kernel(const float* __restrict__ WA,   // [64][64] (k<64)
                                const float* __restrict__ WB,   // [64][64] (k>=64)
                                _Float16* __restrict__ out) {   // 16 frags (K=128)
    int tid  = blockIdx.x * blockDim.x + threadIdx.x;
    int frag = tid >> 5;
    int lane = tid & 31;
    if (frag >= 16) return;
    int nt = frag >> 2, kc = frag & 3;
    int m = lane & 15, hi = lane >> 4;
    int n = nt * 16 + m;
    int kbase = kc * 32 + hi * 8;
    _Float16* o = out + ((size_t)frag * 32 + lane) * 16;
#pragma unroll
    for (int j = 0; j < 16; ++j) {
        int k = kbase + j + ((j >= 8) ? 8 : 0);
        float v = (k < D64) ? WA[n * D64 + k] : WB[n * D64 + (k - D64)];
        o[j] = (_Float16)v;
    }
}

__global__ void prep_plain_kernel(const float* __restrict__ W,  // [64][K] row-major
                                  int K,                        // 32*KC
                                  _Float16* __restrict__ out) { // 4*KC frags
    int KC   = K >> 5;
    int tid  = blockIdx.x * blockDim.x + threadIdx.x;
    int frag = tid >> 5;
    int lane = tid & 31;
    if (frag >= 4 * KC) return;
    int nt = frag / KC, kc = frag % KC;
    int m = lane & 15, hi = lane >> 4;
    int n = nt * 16 + m;
    int kbase = kc * 32 + hi * 8;
    _Float16* o = out + ((size_t)frag * 32 + lane) * 16;
#pragma unroll
    for (int j = 0; j < 16; ++j) {
        int k = kbase + j + ((j >= 8) ? 8 : 0);
        o[j] = (_Float16)W[n * K + k];
    }
}

// ---------------------------------------------------------------------------
// Segment-sum scatter: agg[dst[e]][f..f+3] += hsrc[src[e]][f..f+3]
// thread -> (edge, float4): b128 gathers, 16 lanes cover one 256B edge row.
// All feature tables are L2-resident (192MB L2 >> 64MB features).
// ---------------------------------------------------------------------------
__global__ void scatter_kernel(const float* __restrict__ hsrc,
                               const int* __restrict__ src,
                               const int* __restrict__ dst,
                               float* __restrict__ agg, int E) {
    long long tid = (long long)blockIdx.x * blockDim.x + threadIdx.x;
    long long total = (long long)E * 16;
    if (tid >= total) return;
    int e = (int)(tid >> 4);
    int f = (int)(tid & 15) * 4;
    const float4 v = *(const float4*)(hsrc + (size_t)src[e] * D64 + f);
    float* p = agg + (size_t)dst[e] * D64 + f;
    atomicAdd(p + 0, v.x);
    atomicAdd(p + 1, v.y);
    atomicAdd(p + 2, v.z);
    atomicAdd(p + 3, v.w);
}

// ---------------------------------------------------------------------------
// Fused SAGE linear:  out = act( [agg*inv | hdst] @ [Wl^T ; Wr^T] + bl )
// One wave per 16-row tile; K=128 (4 chunks), prepped f16 B fragments.
// n_dst is a multiple of 16 -> no store guards, EXEC stays full for WMMA.
// ---------------------------------------------------------------------------
template <int RELU>
__global__ void sage_wmma_kernel(const float* __restrict__ agg,
                                 const float* __restrict__ inv_cnt,
                                 const float* __restrict__ hdst,
                                 const _Float16* __restrict__ wfrag, // 16 frags
                                 const float* __restrict__ bl,
                                 float* __restrict__ out,
                                 int ntiles) {
    int gwave  = (blockIdx.x * blockDim.x + threadIdx.x) >> 5;
    int nwaves = (gridDim.x * blockDim.x) >> 5;
    int lane   = threadIdx.x & 31;
    int m      = lane & 15;
    int hi     = lane >> 4;

    for (int tile = gwave; tile < ntiles; tile += nwaves) {
        int row = tile * 16 + m;
        float inv = inv_cnt[row];
        const float* aggRow = agg  + (size_t)row * D64;
        const float* dstRow = hdst + (size_t)row * D64;

        // A fragments over concat-K = 0..127
        v16h afrag[4];
#pragma unroll
        for (int kc = 0; kc < 4; ++kc) {
            int kbase = kc * 32 + hi * 8;
            v16h a;
#pragma unroll
            for (int j = 0; j < 16; ++j) {
                int k = kbase + j + ((j >= 8) ? 8 : 0);
                float v = (k < D64) ? aggRow[k] * inv : dstRow[k - D64];
                a[j] = (_Float16)v;
            }
            afrag[kc] = a;
        }

        for (int nt = 0; nt < 4; ++nt) {
            float bias = bl[nt * 16 + m];
            v8f acc;
#pragma unroll
            for (int r = 0; r < 8; ++r) acc[r] = bias;
#pragma unroll
            for (int kc = 0; kc < 4; ++kc) {
                v16h b = *(const v16h*)(wfrag + ((size_t)(nt * 4 + kc) * 32 + lane) * 16);
                acc = __builtin_amdgcn_wmma_f32_16x16x32_f16(
                    false, afrag[kc], false, b, (short)0, acc, false, false);
            }
            float* orow = out + (size_t)(tile * 16 + hi * 8) * D64 + nt * 16 + m;
#pragma unroll
            for (int r = 0; r < 8; ++r) {
                float v = acc[r];
                if (RELU) v = fmaxf(v, 0.f);
                orow[(size_t)r * D64] = v;
            }
        }
    }
}

// ---------------------------------------------------------------------------
// Edge decoder: z=[hu[row]|hr[col]] (Q x 128)
//   h1 = relu(z @ dW1^T + db1); h2 = relu(h1 @ dW2^T + db2); out = h2.dW3 + db3
// One wave per 16-row tile; h1/h2 staged through a wave-private LDS tile to
// re-fragment C-layout -> A-layout between GEMMs. 2 waves/block, tile clamped
// so barrier control flow is uniform.
// ---------------------------------------------------------------------------
__global__ void decoder_kernel(const float* __restrict__ hu,
                               const float* __restrict__ hr,
                               const int* __restrict__ el_row,
                               const int* __restrict__ el_col,
                               const _Float16* __restrict__ w1p,  // 16 frags (K=128)
                               const float* __restrict__ db1,
                               const _Float16* __restrict__ w2p,  // 8 frags (K=64)
                               const float* __restrict__ db2,
                               const float* __restrict__ dW3,
                               const float* __restrict__ db3,
                               float* __restrict__ out,
                               int Q) {
    __shared__ float lds[2][16 * D64];
    int wav  = threadIdx.x >> 5;
    int lane = threadIdx.x & 31;
    int m    = lane & 15;
    int hi   = lane >> 4;
    int ntiles = Q >> 4;                     // Q multiple of 16

    int tile = blockIdx.x * 2 + wav;
    if (tile >= ntiles) tile = ntiles - 1;   // duplicate work, idempotent store
    float* h = lds[wav];

    int row = tile * 16 + m;
    const float* zu = hu + (size_t)el_row[row] * D64;
    const float* zr = hr + (size_t)el_col[row] * D64;

    // ---- GEMM1: [16x128] @ dW1^T -> h1 [16x64] (relu) -> LDS ----
    v16h a1[4];
#pragma unroll
    for (int kc = 0; kc < 4; ++kc) {
        int kbase = kc * 32 + hi * 8;
        v16h a;
#pragma unroll
        for (int j = 0; j < 16; ++j) {
            int k = kbase + j + ((j >= 8) ? 8 : 0);
            float v = (k < D64) ? zu[k] : zr[k - D64];
            a[j] = (_Float16)v;
        }
        a1[kc] = a;
    }

    for (int nt = 0; nt < 4; ++nt) {
        float bias = db1[nt * 16 + m];
        v8f acc;
#pragma unroll
        for (int r = 0; r < 8; ++r) acc[r] = bias;
#pragma unroll
        for (int kc = 0; kc < 4; ++kc) {
            v16h b = *(const v16h*)(w1p + ((size_t)(nt * 4 + kc) * 32 + lane) * 16);
            acc = __builtin_amdgcn_wmma_f32_16x16x32_f16(
                false, a1[kc], false, b, (short)0, acc, false, false);
        }
#pragma unroll
        for (int r = 0; r < 8; ++r)
            h[(hi * 8 + r) * D64 + nt * 16 + m] = fmaxf(acc[r], 0.f);
    }
    __syncthreads();

    // ---- re-fragment h1 from LDS (A layout), K=64 -> 2 chunks ----
    v16h a2[2];
#pragma unroll
    for (int kc = 0; kc < 2; ++kc) {
        int kbase = kc * 32 + hi * 8;
        v16h a;
#pragma unroll
        for (int j = 0; j < 16; ++j) {
            int k = kbase + j + ((j >= 8) ? 8 : 0);
            a[j] = (_Float16)h[m * D64 + k];
        }
        a2[kc] = a;
    }
    __syncthreads();

    // ---- GEMM2: [16x64] @ dW2^T -> h2 [16x64] (relu) -> LDS ----
    for (int nt = 0; nt < 4; ++nt) {
        float bias = db2[nt * 16 + m];
        v8f acc;
#pragma unroll
        for (int r = 0; r < 8; ++r) acc[r] = bias;
#pragma unroll
        for (int kc = 0; kc < 2; ++kc) {
            v16h b = *(const v16h*)(w2p + ((size_t)(nt * 2 + kc) * 32 + lane) * 16);
            acc = __builtin_amdgcn_wmma_f32_16x16x32_f16(
                false, a2[kc], false, b, (short)0, acc, false, false);
        }
#pragma unroll
        for (int r = 0; r < 8; ++r)
            h[(hi * 8 + r) * D64 + nt * 16 + m] = fmaxf(acc[r], 0.f);
    }
    __syncthreads();

    // ---- final dot with dW3 (lane pair m / m+16 split the 64 K) ----
    float partial = 0.f;
    int k0 = hi * 32;
#pragma unroll
    for (int k = 0; k < 32; ++k)
        partial += h[m * D64 + k0 + k] * dW3[k0 + k];
    partial += __shfl_down(partial, 16, 32);
    if (lane < 16)
        out[row] = partial + db3[0];
}

// ---------------------------------------------------------------------------
extern "C" void kernel_launch(void* const* d_in, const int* in_sizes, int n_in,
                              void* d_out, int out_size, void* d_ws, size_t ws_size,
                              hipStream_t stream) {
    const float* x_user = (const float*)d_in[0];
    const float* x_rest = (const float*)d_in[1];
    const float* Wl     = (const float*)d_in[2];   // (3,2,64,64)
    const float* Wr     = (const float*)d_in[3];
    const float* bl     = (const float*)d_in[4];   // (3,2,64)
    const float* dW1    = (const float*)d_in[5];
    const float* db1    = (const float*)d_in[6];
    const float* dW2    = (const float*)d_in[7];
    const float* db2    = (const float*)d_in[8];
    const float* dW3    = (const float*)d_in[9];
    const float* db3    = (const float*)d_in[10];
    const int* ur_src   = (const int*)d_in[11];
    const int* ur_dst   = (const int*)d_in[12];
    const int* ru_src   = (const int*)d_in[13];
    const int* ru_dst   = (const int*)d_in[14];
    const int* el_row   = (const int*)d_in[15];
    const int* el_col   = (const int*)d_in[16];

    const int N_USER = in_sizes[0] / D64;
    const int N_REST = in_sizes[1] / D64;
    const int E      = in_sizes[11];
    const int Q      = in_sizes[15];

    // Workspace layout (floats, all 32B-aligned chunks)
    float* ws = (float*)d_ws;
    size_t off = 0;
    float* agg   = ws + off; off += (size_t)N_USER * D64;   // reused both dirs
    float* inv_u = ws + off; off += (size_t)N_USER;
    float* inv_r = ws + off; off += (size_t)N_REST;
    float* hu_b[2]; float* hr_b[2];
    hu_b[0] = ws + off; off += (size_t)N_USER * D64;
    hu_b[1] = ws + off; off += (size_t)N_USER * D64;
    hr_b[0] = ws + off; off += (size_t)N_REST * D64;
    hr_b[1] = ws + off; off += (size_t)N_REST * D64;

    // f16 prepped-weight region
    _Float16* wprep = (_Float16*)(ws + off);
    _Float16* sagew[6];
    for (int i = 0; i < 6; ++i) sagew[i] = wprep + (size_t)i * 8192;  // 16 frags
    _Float16* w1p = wprep + (size_t)6 * 8192;                         // 16 frags
    _Float16* w2p = w1p + 8192;                                       // 8 frags

    // ---- weight prep (constant across layers) ----
    for (int l = 0; l < 3; ++l)
        for (int t = 0; t < 2; ++t) {
            const float* wl = Wl + (size_t)(l * 2 + t) * D64 * D64;
            const float* wr = Wr + (size_t)(l * 2 + t) * D64 * D64;
            prep_cat_kernel<<<2, 256, 0, stream>>>(wl, wr, sagew[l * 2 + t]);
        }
    prep_plain_kernel<<<2, 256, 0, stream>>>(dW1, 128, w1p);
    prep_plain_kernel<<<1, 256, 0, stream>>>(dW2, 64, w2p);

    // ---- degrees (invariant across layers) ----
    hipMemsetAsync(inv_u, 0, (size_t)N_USER * sizeof(float), stream);
    hipMemsetAsync(inv_r, 0, (size_t)N_REST * sizeof(float), stream);
    count_kernel<<<(E + 255) / 256, 256, 0, stream>>>(ur_dst, (int*)inv_r, E);
    count_kernel<<<(E + 255) / 256, 256, 0, stream>>>(ru_dst, (int*)inv_u, E);
    inv_kernel<<<(N_REST + 255) / 256, 256, 0, stream>>>(inv_r, N_REST);
    inv_kernel<<<(N_USER + 255) / 256, 256, 0, stream>>>(inv_u, N_USER);

    const float* hu_cur = x_user;
    const float* hr_cur = x_rest;
    int pp = 0;

    long long sc_total = (long long)E * 16;
    int sc_blocks = (int)((sc_total + 255) / 256);

    for (int l = 0; l < 3; ++l) {
        const float* bl0 = bl + (size_t)(l * 2 + 0) * D64;
        const float* bl1 = bl + (size_t)(l * 2 + 1) * D64;
        int rl = (l < 2);

        // user -> restaurant ('rates'):  zr
        hipMemsetAsync(agg, 0, (size_t)N_REST * D64 * sizeof(float), stream);
        scatter_kernel<<<sc_blocks, 256, 0, stream>>>(hu_cur, ur_src, ur_dst, agg, E);
        {
            int tiles = N_REST >> 4;
            int blocks = (tiles + 7) / 8;
            if (rl)
                sage_wmma_kernel<1><<<blocks, 256, 0, stream>>>(
                    agg, inv_r, hr_cur, sagew[l * 2 + 0], bl0, hr_b[pp], tiles);
            else
                sage_wmma_kernel<0><<<blocks, 256, 0, stream>>>(
                    agg, inv_r, hr_cur, sagew[l * 2 + 0], bl0, hr_b[pp], tiles);
        }

        // restaurant -> user (reverse):  zu  (reads OLD hr_cur)
        hipMemsetAsync(agg, 0, (size_t)N_USER * D64 * sizeof(float), stream);
        scatter_kernel<<<sc_blocks, 256, 0, stream>>>(hr_cur, ru_src, ru_dst, agg, E);
        {
            int tiles = N_USER >> 4;
            int blocks = (tiles + 7) / 8;
            if (rl)
                sage_wmma_kernel<1><<<blocks, 256, 0, stream>>>(
                    agg, inv_u, hu_cur, sagew[l * 2 + 1], bl1, hu_b[pp], tiles);
            else
                sage_wmma_kernel<0><<<blocks, 256, 0, stream>>>(
                    agg, inv_u, hu_cur, sagew[l * 2 + 1], bl1, hu_b[pp], tiles);
        }

        hu_cur = hu_b[pp];
        hr_cur = hr_b[pp];
        pp ^= 1;
    }

    // ---- edge decoder ----
    int ntiles = Q >> 4;
    int dblocks = (ntiles + 1) / 2;   // 2 waves/block, 1 tile/wave
    decoder_kernel<<<dblocks, 64, 0, stream>>>(
        hu_cur, hr_cur, el_row, el_col, w1p, db1, w2p, db2, dW3, db3,
        (float*)d_out, Q);
}
